// HyperbolicGraphConv_30124900614684
// MI455X (gfx1250) — matrix operand order repaired and checked by reading, not compiled, not run on hardware
//
#include <hip/hip_runtime.h>

// ---------------------------------------------------------------------------
// HyperbolicGraphConv for MI455X (gfx1250, wave32)
//   tangent = artanh(||x||) x/||x||          (f32 in, bf16 out)
//   h       = tangent @ W^T + b              (bf16 WMMA 16x16x32, f32 acc)
//   sums    = segment_sum(h[src], dst)       (f32 global atomics into d_out)
//   out     = expmap0(sums / max(deg,1))     (in-place finalize)
// ---------------------------------------------------------------------------

#define DD 512
#define EPSF 1e-7f

typedef __attribute__((ext_vector_type(16))) __bf16 v16bf;
typedef __attribute__((ext_vector_type(8)))  __bf16 v8bf;
typedef __attribute__((ext_vector_type(8)))  float  v8f;

static __device__ __forceinline__ __bf16 f2bf(float f) {
    unsigned u = __builtin_bit_cast(unsigned, f);
    u += 0x7FFFu + ((u >> 16) & 1u);               // round-to-nearest-even
    return __builtin_bit_cast(__bf16, (unsigned short)(u >> 16));
}
static __device__ __forceinline__ float bfbits2f(unsigned lo16) {
    return __builtin_bit_cast(float, lo16 << 16);
}

// ---------------- zero accumulators (d_out sums + degree) -------------------
__global__ void zero_kernel(float* __restrict__ out, float* __restrict__ deg,
                            int n512, int N) {
    int stride = gridDim.x * blockDim.x;
    int i = blockIdx.x * blockDim.x + threadIdx.x;
    float4 z = {0.f, 0.f, 0.f, 0.f};
    for (int j = i; j < (n512 >> 2); j += stride) ((float4*)out)[j] = z;
    for (int j = i; j < N; j += stride) deg[j] = 0.f;
}

// ---------------- logmap0 + f32->bf16 (one block of 128 per row) ------------
__global__ void __launch_bounds__(128)
logmap_kernel(const float* __restrict__ x, __bf16* __restrict__ t, int N) {
    int row = blockIdx.x;
    int tid = threadIdx.x;                       // 128 threads * float4 = 512
    float4 v = ((const float4*)(x + (size_t)row * DD))[tid];
    float s = v.x * v.x + v.y * v.y + v.z * v.z + v.w * v.w;
    __shared__ float red[128];
    red[tid] = s;
    __syncthreads();
    for (int off = 64; off > 0; off >>= 1) {
        if (tid < off) red[tid] += red[tid + off];
        __syncthreads();
    }
    float n  = sqrtf(red[0]);
    float nc = fminf(fmaxf(n, EPSF), 1.f - EPSF);
    float sc = atanhf(nc) / nc;
    __bf16* tr = t + (size_t)row * DD + tid * 4;
    tr[0] = f2bf(v.x * sc); tr[1] = f2bf(v.y * sc);
    tr[2] = f2bf(v.z * sc); tr[3] = f2bf(v.w * sc);
}

// ---------------- W f32 -> bf16 --------------------------------------------
__global__ void convw_kernel(const float* __restrict__ W, __bf16* __restrict__ Wb) {
    int i = blockIdx.x * blockDim.x + threadIdx.x;   // 262144 elements
    Wb[i] = f2bf(W[i]);
}

// ---------------- GEMM: h = tangent @ W^T + b  (bf16 WMMA, f32 acc) ---------
// Wave tile 16(M) x 128(N); block = 8 waves stacked in M -> 128x128 tile.
// Single per-lane base pointer per stream; nt/k0 contributions are uniform
// unroll constants folded into the 24-bit instruction offset.
// grid = (ceil(N/128), 512/128)
__global__ void __launch_bounds__(256)
gemm_kernel(const __bf16* __restrict__ A,   // [N,512] tangent, row-major
            const __bf16* __restrict__ Wb,  // [512,512] W, row-major (= B col-major)
            const float*  __restrict__ bias,
            __bf16* __restrict__ H,         // [N,512] out
            int N) {
    const int lane   = threadIdx.x & 31;
    const int wave   = threadIdx.x >> 5;
    const int l15    = lane & 15;
    const int kh     = lane >> 4;              // 0 | 1 (half-wave)
    const int m0     = blockIdx.x * 128 + wave * 16;
    const int n0     = blockIdx.y * 128;

    // clamp A-row for loads (WMMA requires EXEC all-ones; guard only stores)
    int mload = m0 + l15;
    if (mload >= N) mload = N - 1;
    // A base: lane holds K = kh*16..+15 contiguous per 32-K step
    const __bf16* aB = A + (size_t)mload * DD + kh * 16;

    // B base: lane = col n0+l15 of tile 0; K = kh*16..+15 contiguous.
    // nt adds a UNIFORM +nt*16*512 elements; k0 adds +k0 -> all immediates.
    const __bf16* bB = Wb + (((size_t)(n0 + l15)) << 9) + kh * 16;

    v8f acc[8];
#pragma unroll
    for (int nt = 0; nt < 8; ++nt) acc[nt] = (v8f)0.f;

#pragma unroll
    for (int k0 = 0; k0 < DD; k0 += 32) {
        v8bf alo = *(const v8bf*)(aB + k0);
        v8bf ahi = *(const v8bf*)(aB + k0 + 8);
        v16bf a  = __builtin_shufflevector(alo, ahi,
                       0,1,2,3,4,5,6,7,8,9,10,11,12,13,14,15);
#pragma unroll
        for (int nt = 0; nt < 8; ++nt) {
            v16bf bb = *(const v16bf*)(bB + nt * 8192 + k0);   // 32B aligned
            acc[nt] = __builtin_amdgcn_wmma_f32_16x16x32_bf16(
                false, a, false, bb, (short)0, acc[nt], false, false);
        }
    }

    // epilogue: C VGPR r -> row m0 + r + 8*kh, col n0 + nt*16 + l15
#pragma unroll
    for (int nt = 0; nt < 8; ++nt) {
        int col = n0 + nt * 16 + l15;
        float bv = bias[col];
#pragma unroll
        for (int r = 0; r < 8; ++r) {
            int row = m0 + r + 8 * kh;
            if (row < N)
                H[(size_t)row * DD + col] = f2bf(acc[nt][r] + bv);
        }
    }
}

// ---------------- edge scatter: one wave per edge ---------------------------
__global__ void __launch_bounds__(256)
scatter_kernel(const __bf16* __restrict__ H, const int* __restrict__ src,
               const int* __restrict__ dst, float* __restrict__ out,
               float* __restrict__ deg, int E) {
    int lane = threadIdx.x & 31;
    int e = blockIdx.x * 8 + (threadIdx.x >> 5);
    if (e >= E) return;
    int s = src[e], d = dst[e];
    const __bf16* hr = H + (size_t)s * DD;
    float* orow = out + (size_t)d * DD;
    if (lane == 0)
        __hip_atomic_fetch_add(&deg[d], 1.0f, __ATOMIC_RELAXED,
                               __HIP_MEMORY_SCOPE_AGENT);
#pragma unroll
    for (int it = 0; it < 4; ++it) {
        int off = it * 128 + lane * 4;                 // 8B coalesced per lane
        uint2 pk = *(const uint2*)(hr + off);
        float f0 = bfbits2f(pk.x & 0xffffu);
        float f1 = bfbits2f(pk.x >> 16);
        float f2 = bfbits2f(pk.y & 0xffffu);
        float f3 = bfbits2f(pk.y >> 16);
        __hip_atomic_fetch_add(orow + off + 0, f0, __ATOMIC_RELAXED, __HIP_MEMORY_SCOPE_AGENT);
        __hip_atomic_fetch_add(orow + off + 1, f1, __ATOMIC_RELAXED, __HIP_MEMORY_SCOPE_AGENT);
        __hip_atomic_fetch_add(orow + off + 2, f2, __ATOMIC_RELAXED, __HIP_MEMORY_SCOPE_AGENT);
        __hip_atomic_fetch_add(orow + off + 3, f3, __ATOMIC_RELAXED, __HIP_MEMORY_SCOPE_AGENT);
    }
}

// ---------------- finalize: mean + expmap0, in place ------------------------
__global__ void __launch_bounds__(128)
expmap_kernel(float* __restrict__ out, const float* __restrict__ deg, int N) {
    int row = blockIdx.x;
    int tid = threadIdx.x;
    float inv = 1.0f / fmaxf(deg[row], 1.0f);
    float4* orow = (float4*)(out + (size_t)row * DD);
    float4 v = orow[tid];
    v.x *= inv; v.y *= inv; v.z *= inv; v.w *= inv;
    float s = v.x * v.x + v.y * v.y + v.z * v.z + v.w * v.w;
    __shared__ float red[128];
    red[tid] = s;
    __syncthreads();
    for (int off = 64; off > 0; off >>= 1) {
        if (tid < off) red[tid] += red[tid + off];
        __syncthreads();
    }
    float n  = fmaxf(sqrtf(red[0]), EPSF);
    float sc = tanhf(n) / n;
    v.x *= sc; v.y *= sc; v.z *= sc; v.w *= sc;
    orow[tid] = v;
}

// ---------------------------------------------------------------------------
extern "C" void kernel_launch(void* const* d_in, const int* in_sizes, int n_in,
                              void* d_out, int out_size, void* d_ws, size_t ws_size,
                              hipStream_t stream) {
    const float* x   = (const float*)d_in[0];
    const int*   src = (const int*)d_in[1];
    const int*   dst = (const int*)d_in[2];
    const float* W   = (const float*)d_in[3];
    const float* b   = (const float*)d_in[4];
    float* out = (float*)d_out;

    const int N = in_sizes[0] / DD;
    const int E = in_sizes[1];

    // workspace layout (bytes, all 256-aligned for these sizes)
    char* ws = (char*)d_ws;
    size_t tanBytes = (size_t)N * DD * sizeof(__bf16);
    __bf16* tangent = (__bf16*)ws;                 ws += tanBytes;
    __bf16* hbf     = (__bf16*)ws;                 ws += tanBytes;
    __bf16* Wbf     = (__bf16*)ws;                 ws += (size_t)DD * DD * sizeof(__bf16);
    float*  deg     = (float*)ws;

    zero_kernel<<<2048, 256, 0, stream>>>(out, deg, N * DD, N);
    logmap_kernel<<<N, 128, 0, stream>>>(x, tangent, N);
    convw_kernel<<<(DD * DD) / 256, 256, 0, stream>>>(W, Wbf);
    gemm_kernel<<<dim3((N + 127) / 128, DD / 128), 256, 0, stream>>>(
        tangent, Wbf, b, hbf, N);
    scatter_kernel<<<(E + 7) / 8, 256, 0, stream>>>(hbf, src, dst, out, deg, E);
    expmap_kernel<<<N, 128, 0, stream>>>(out, deg, N);
}
